// FrequencyTimeAttention_65412351918497
// MI455X (gfx1250) — compile-verified, matrix operand order checked by reading, not run on hardware
//
#include <hip/hip_runtime.h>

// ---------------------------------------------------------------------------
// MI455X / gfx1250 fused frequency-time attention block.
// GEMM-shaped math on v_wmma_f32_16x16x32_bf16 (wave32), fp32 accumulation.
// ALL tile staging (GEMMs + attention K/V) uses CDNA5 async global->LDS
// copies (ASYNCcnt), double-buffered so the DMA overlaps WMMA/softmax.
// ---------------------------------------------------------------------------

typedef __bf16 bf16_t;
typedef __attribute__((ext_vector_type(16))) __bf16 v16bf;
typedef __attribute__((ext_vector_type(8)))  float  v8f;

#define BATCH 4
#define CDIM  512
#define SEQ   2048          // FDIM*TDIM
#define NTOK  8192          // BATCH*SEQ
#define HEADS 8
#define DHEAD 64
#define N_QKV 1536

union FragBF { v16bf v; uint4 q[2]; };

// 16x32 bf16 WMMA operand loader (A layout; B mirrored with lane->column,
// so B tiles are staged transposed [N][K] row-major). Works on LDS or global
// pointers; rows must be 16-byte aligned (ld % 8 == 0).
__device__ __forceinline__ v16bf ld_frag(const bf16_t* p, int ld) {
  const int lane = threadIdx.x & 31;
  const int r  = lane & 15;
  const int kb = (lane >> 4) << 3;      // 0 or 8
  const bf16_t* row = p + r * ld + kb;
  FragBF f;
  f.q[0] = *(const uint4*)(row);        // K = kb .. kb+7
  f.q[1] = *(const uint4*)(row + 16);   // K = kb+16 .. kb+23
  return f.v;
}

__device__ __forceinline__ v8f wmma_bf16(v16bf a, v16bf b, v8f c) {
  return __builtin_amdgcn_wmma_f32_16x16x32_bf16(
      false, a, false, b, (short)0, c, false, false);
}

__device__ __forceinline__ v8f v8f_zero() {
  v8f z = {0.f, 0.f, 0.f, 0.f, 0.f, 0.f, 0.f, 0.f};
  return z;
}

// CDNA5 async 16-byte global -> LDS copy (tracked with ASYNCcnt).
__device__ __forceinline__ void async_ld16(const bf16_t* lds_dst, const bf16_t* gsrc) {
  uint32_t loff = (uint32_t)(uintptr_t)lds_dst;   // low 32 bits = LDS offset
  asm volatile("global_load_async_to_lds_b128 %0, %1, off"
               :: "v"(loff), "v"(gsrc) : "memory");
}
__device__ __forceinline__ void wait_async0() {
  asm volatile("s_wait_asynccnt 0" ::: "memory");
}
__device__ __forceinline__ void wait_async4() {   // allow next tile in flight
  asm volatile("s_wait_asynccnt 4" ::: "memory");
}

// ---------------------------------------------------------------------------
// Weight prep (one-shot): fp32 (K,N) row-major -> bf16 (N,K) row-major.
// ---------------------------------------------------------------------------
__global__ __launch_bounds__(256) void k_wprep(const float* __restrict__ w,
                                               bf16_t* __restrict__ wT,
                                               int K, int N) {
  for (int i = blockIdx.x * 256 + threadIdx.x; i < K * N; i += gridDim.x * 256) {
    int k = i / N, n = i - k * N;
    wT[(size_t)n * K + k] = (bf16_t)w[i];
  }
}

// ---------------------------------------------------------------------------
// Kernel 1: LayerNorm over c of x (b,c,f,t) -> xn bf16 (token, c).
// ---------------------------------------------------------------------------
__global__ __launch_bounds__(256) void k_ln1(const float* __restrict__ x,
                                             const float* __restrict__ gamma,
                                             const float* __restrict__ beta,
                                             bf16_t* __restrict__ xn) {
  __shared__ float s_sum[2][128];
  __shared__ float s_sq[2][128];
  __shared__ float s_mu[128];
  __shared__ float s_rs[128];
  const int tid  = threadIdx.x;
  const int tl   = tid & 127;
  const int half = tid >> 7;
  const int c0   = half * 256;
  const int token = blockIdx.x * 128 + tl;
  const int bb = token >> 11;
  const int st = token & 2047;
  const float* xb = x + (size_t)bb * CDIM * SEQ + st;

  float sum = 0.f, sq = 0.f;
  for (int cc = c0; cc < c0 + 256; cc++) {
    float v = xb[(size_t)cc * SEQ];
    sum += v; sq += v * v;
  }
  s_sum[half][tl] = sum;
  s_sq[half][tl]  = sq;
  __syncthreads();
  if (tid < 128) {
    float s  = s_sum[0][tid] + s_sum[1][tid];
    float q  = s_sq[0][tid]  + s_sq[1][tid];
    float mu = s * (1.f / CDIM);
    float var = q * (1.f / CDIM) - mu * mu;
    s_mu[tid] = mu;
    s_rs[tid] = rsqrtf(var + 1e-5f);
  }
  __syncthreads();
  const float mu = s_mu[tl], rs = s_rs[tl];
  bf16_t* xr = xn + (size_t)token * CDIM;
  for (int cc = c0; cc < c0 + 256; cc += 2) {
    float v0 = xb[(size_t)cc * SEQ];
    float v1 = xb[(size_t)(cc + 1) * SEQ];
    union { bf16_t h[2]; uint32_t u; } pk;
    pk.h[0] = (bf16_t)((v0 - mu) * rs * gamma[cc] + beta[cc]);
    pk.h[1] = (bf16_t)((v1 - mu) * rs * gamma[cc + 1] + beta[cc + 1]);
    *(uint32_t*)(xr + cc) = pk.u;
  }
}

// ---------------------------------------------------------------------------
// Shared GEMM core: C(128x128) = A(128x512 bf16) @ B^T(128 rows x 512 bf16),
// async double-buffered LDS staging, 8 waves x 16 rows, bf16 WMMA.
// ---------------------------------------------------------------------------
template <typename EpilogueFn>
__device__ __forceinline__ void gemm_core(const bf16_t* __restrict__ A,
                                          const bf16_t* __restrict__ BT,
                                          int m0, int n0,
                                          bf16_t* As /*[2][128*48]*/,
                                          bf16_t* Bs /*[2][128*48]*/,
                                          EpilogueFn epi) {
  const int tid  = threadIdx.x;
  const int wv   = tid >> 5;
  const int r0 = tid >> 2, r1 = r0 + 64, q4 = tid & 3;   // staging chunk coords

  // 4 async instructions per wave per k-tile (2 A rows + 2 B rows per thread)
  auto stage = [&](int buf, int k0) {
    bf16_t* a = As + buf * (128 * 48);
    bf16_t* b = Bs + buf * (128 * 48);
    async_ld16(a + r0 * 48 + q4 * 8, A  + (size_t)(m0 + r0) * CDIM + k0 + q4 * 8);
    async_ld16(a + r1 * 48 + q4 * 8, A  + (size_t)(m0 + r1) * CDIM + k0 + q4 * 8);
    async_ld16(b + r0 * 48 + q4 * 8, BT + (size_t)(n0 + r0) * CDIM + k0 + q4 * 8);
    async_ld16(b + r1 * 48 + q4 * 8, BT + (size_t)(n0 + r1) * CDIM + k0 + q4 * 8);
  };

  v8f acc[8];
#pragma unroll
  for (int i = 0; i < 8; i++) acc[i] = v8f_zero();

  stage(0, 0);
  for (int it = 0; it < 16; it++) {
    const int cur = it & 1;
    if (it + 1 < 16) {
      stage(cur ^ 1, (it + 1) * 32);   // next tile's DMA overlaps compute
      wait_async4();                   // tile it complete, it+1 in flight
    } else {
      wait_async0();
    }
    __syncthreads();                   // tile it visible to all waves

    const bf16_t* ac = As + cur * (128 * 48);
    const bf16_t* bc = Bs + cur * (128 * 48);
    v16bf a = ld_frag(ac + (wv * 16) * 48, 48);
#pragma unroll
    for (int g = 0; g < 2; g++) {      // two groups of 4 keeps VGPR pressure low
      v16bf b[4];
#pragma unroll
      for (int i = 0; i < 4; i++)
        b[i] = ld_frag(bc + ((g * 4 + i) * 16) * 48, 48);
#pragma unroll
      for (int i = 0; i < 4; i++)
        acc[g * 4 + i] = wmma_bf16(a, b[i], acc[g * 4 + i]);
    }
    __syncthreads();                   // reads retired before buffer reuse
  }
  epi(acc);
}

// ---------------------------------------------------------------------------
// Kernel 2: QKV projection. K pre-scaled by dh^-0.5; V written d-major.
// ---------------------------------------------------------------------------
__global__ __launch_bounds__(256) void k_qkv(const bf16_t* __restrict__ xn,
                                             const bf16_t* __restrict__ wT,
                                             const float* __restrict__ bias,
                                             bf16_t* __restrict__ qh,
                                             bf16_t* __restrict__ kh,
                                             bf16_t* __restrict__ vT) {
  __shared__ __align__(16) bf16_t As[2][128 * 48];
  __shared__ __align__(16) bf16_t Bs[2][128 * 48];
  const int tid  = threadIdx.x;
  const int wv   = tid >> 5;
  const int lane = tid & 31;
  const int hi = lane >> 4, nl = lane & 15;
  const int m0 = blockIdx.x * 128;
  const int n0 = blockIdx.y * 128;

  gemm_core(xn, wT, m0, n0, As[0], Bs[0], [&](v8f (&acc)[8]) {
    const int which = n0 >> 9;                        // uniform per block
    const float scl = (which == 1) ? 0.125f : 1.0f;   // fold dh^-0.5 into K
    const int token0 = m0 + wv * 16 + 8 * hi;         // 8 consecutive tokens
    const int bb = token0 >> 11, ss0 = token0 & 2047;
#pragma unroll
    for (int ns = 0; ns < 8; ns++) {
      const int n = n0 + ns * 16 + nl;
      const float bv = bias[n];
      const int cc = n & 511;
      const int head = cc >> 6;
      const int d = cc & 63;
      const size_t bhh = (size_t)(bb * HEADS + head);
      if (which == 2) {
        // V: d-major rows contiguous over tokens -> one b128 store
        union { bf16_t h[8]; uint4 u; } pk;
#pragma unroll
        for (int j = 0; j < 8; j++) pk.h[j] = (bf16_t)(acc[ns][j] + bv);
        *(uint4*)(vT + (bhh * DHEAD + d) * SEQ + ss0) = pk.u;
      } else {
        bf16_t* dst = (which == 0) ? qh : kh;
#pragma unroll
        for (int j = 0; j < 8; j++)
          dst[(bhh * SEQ + ss0 + j) * DHEAD + d] = (bf16_t)((acc[ns][j] + bv) * scl);
      }
    }
  });
}

// ---------------------------------------------------------------------------
// Kernel 3: flash attention per (b*h); 128 queries/block, KV tiles of 64,
// double-buffered async global->LDS staging overlapping compute.
// ---------------------------------------------------------------------------
__global__ __launch_bounds__(256) void k_attn(const bf16_t* __restrict__ qh,
                                              const bf16_t* __restrict__ kh,
                                              const bf16_t* __restrict__ vT,
                                              float* __restrict__ o) {
  __shared__ __align__(16) bf16_t Ks[2][64 * 72];    // [kv][d]  (B of Q.K^T)
  __shared__ __align__(16) bf16_t Vt[2][64 * 72];    // [d][kv]  (B of P.V)
  __shared__ __align__(16) bf16_t Ps[8 * 16 * 72];   // per-wave P

  const int tid  = threadIdx.x;
  const int wv   = tid >> 5;
  const int lane = tid & 31;
  const int hi = lane >> 4, nl = lane & 15;
  const int qt = blockIdx.x;
  const int bh = blockIdx.y;
  const size_t base = (size_t)bh * SEQ * DHEAD;

  const int r0 = tid >> 3, qq = tid & 7;   // staging chunk coords
  const bf16_t* kb0 = kh + base;
  const bf16_t* vb0 = vT + base;
  auto stage_tile = [&](int buf, int kv0) {   // 4 async instrs / wave / tile
    const bf16_t* ksrc = kb0 + (size_t)kv0 * DHEAD;
    const bf16_t* vsrc = vb0 + kv0;
    async_ld16(&Ks[buf][r0 * 72 + qq * 8],        ksrc + (size_t)r0 * DHEAD + qq * 8);
    async_ld16(&Ks[buf][(r0 + 32) * 72 + qq * 8], ksrc + (size_t)(r0 + 32) * DHEAD + qq * 8);
    async_ld16(&Vt[buf][r0 * 72 + qq * 8],        vsrc + (size_t)r0 * SEQ + qq * 8);
    async_ld16(&Vt[buf][(r0 + 32) * 72 + qq * 8], vsrc + (size_t)(r0 + 32) * SEQ + qq * 8);
  };

  stage_tile(0, 0);

  // Per-wave Q fragments directly from global (rows are 128B, 16B aligned).
  const bf16_t* qrow = qh + base + (size_t)(qt * 128 + wv * 16) * DHEAD;
  const v16bf aq0 = ld_frag(qrow, DHEAD);
  const v16bf aq1 = ld_frag(qrow + 32, DHEAD);

  v8f Oacc[4];
  float m8[8], l8[8];
#pragma unroll
  for (int i = 0; i < 4; i++) Oacc[i] = v8f_zero();
#pragma unroll
  for (int j = 0; j < 8; j++) { m8[j] = -3.0e38f; l8[j] = 0.f; }

  bf16_t* Pw = Ps + (wv * 16) * 72;

  for (int kt = 0; kt < SEQ / 64; kt++) {
    const int cur = kt & 1;
    if (kt + 1 < SEQ / 64) {
      stage_tile(cur ^ 1, (kt + 1) * 64);   // overlap next tile's DMA
      if (kt + 2 < SEQ / 64)                // and nudge kt+2 toward L2
        __builtin_prefetch(kb0 + (size_t)(kt + 2) * 64 * DHEAD, 0, 1);
      wait_async4();                        // tile kt complete, kt+1 in flight
    } else {
      wait_async0();
    }
    __syncthreads();                        // tile kt visible to all waves

    // scores S(16x64) = Qw @ K^T (scale pre-folded into K); pairs of frags
    v8f s[4];
#pragma unroll
    for (int ns = 0; ns < 4; ns++) {
      v16bf bk0 = ld_frag(Ks[cur] + (ns * 16) * 72, 72);
      v16bf bk1 = ld_frag(Ks[cur] + (ns * 16) * 72 + 32, 72);
      v8f z = v8f_zero();
      z = wmma_bf16(aq0, bk0, z);
      s[ns] = wmma_bf16(aq1, bk1, z);
    }

    // online softmax; element (M,N): VGPR j -> row j + 8*hi
#pragma unroll
    for (int j = 0; j < 8; j++) {
      float mt = fmaxf(fmaxf(s[0][j], s[1][j]), fmaxf(s[2][j], s[3][j]));
      mt = fmaxf(mt, __shfl_xor(mt, 1, 32));
      mt = fmaxf(mt, __shfl_xor(mt, 2, 32));
      mt = fmaxf(mt, __shfl_xor(mt, 4, 32));
      mt = fmaxf(mt, __shfl_xor(mt, 8, 32));
      const float mn = fmaxf(m8[j], mt);
      const float alpha = __expf(m8[j] - mn);
      const float p0 = __expf(s[0][j] - mn);
      const float p1 = __expf(s[1][j] - mn);
      const float p2 = __expf(s[2][j] - mn);
      const float p3 = __expf(s[3][j] - mn);
      float ps = p0 + p1 + p2 + p3;
      ps += __shfl_xor(ps, 1, 32);
      ps += __shfl_xor(ps, 2, 32);
      ps += __shfl_xor(ps, 4, 32);
      ps += __shfl_xor(ps, 8, 32);
      l8[j] = l8[j] * alpha + ps;
      m8[j] = mn;
#pragma unroll
      for (int nd = 0; nd < 4; nd++) Oacc[nd][j] *= alpha;
      const int row = j + 8 * hi;
      Pw[row * 72 +  0 + nl] = (bf16_t)p0;
      Pw[row * 72 + 16 + nl] = (bf16_t)p1;
      Pw[row * 72 + 32 + nl] = (bf16_t)p2;
      Pw[row * 72 + 48 + nl] = (bf16_t)p3;
    }
    __syncthreads();   // P visible (Ks[cur] reads also retired)

    // O += P(16x64) @ V(64x64)
#pragma unroll
    for (int kc = 0; kc < 2; kc++) {
      v16bf ap = ld_frag(Pw + kc * 32, 72);
#pragma unroll
      for (int nd = 0; nd < 4; nd++) {
        v16bf bv = ld_frag(Vt[cur] + (nd * 16) * 72 + kc * 32, 72);
        Oacc[nd] = wmma_bf16(ap, bv, Oacc[nd]);
      }
    }
    __syncthreads();   // Vt[cur]/Ps reads done; safe to overwrite next iter
  }

  const int bb = bh >> 3, head = bh & 7;
#pragma unroll
  for (int j = 0; j < 8; j++) {
    const float inv = 1.f / l8[j];
    const int ss = qt * 128 + wv * 16 + j + 8 * hi;
    float* orow = o + ((size_t)bb * SEQ + ss) * CDIM + head * DHEAD;
#pragma unroll
    for (int nd = 0; nd < 4; nd++)
      orow[nd * 16 + nl] = Oacc[nd][j] * inv;
  }
}

// ---------------------------------------------------------------------------
// Kernel 4: LayerNorm2 over c (token-contiguous fp32 -> bf16 (token, c)).
// ---------------------------------------------------------------------------
__global__ __launch_bounds__(256) void k_ln2(const float* __restrict__ o,
                                             const float* __restrict__ gamma,
                                             const float* __restrict__ beta,
                                             bf16_t* __restrict__ on) {
  const int token = blockIdx.x * 256 + threadIdx.x;
  const float* r = o + (size_t)token * CDIM;
  float sum = 0.f, sq = 0.f;
  for (int c = 0; c < CDIM; c += 4) {
    float4 v = *(const float4*)(r + c);
    sum += v.x + v.y + v.z + v.w;
    sq  += v.x * v.x + v.y * v.y + v.z * v.z + v.w * v.w;
  }
  const float mu = sum * (1.f / CDIM);
  const float rs = rsqrtf(sq * (1.f / CDIM) - mu * mu + 1e-5f);
  bf16_t* w = on + (size_t)token * CDIM;
  for (int c = 0; c < CDIM; c += 4) {
    float4 v = *(const float4*)(r + c);
    union { bf16_t h[4]; uint2 u; } pk;
    pk.h[0] = (bf16_t)((v.x - mu) * rs * gamma[c]     + beta[c]);
    pk.h[1] = (bf16_t)((v.y - mu) * rs * gamma[c + 1] + beta[c + 1]);
    pk.h[2] = (bf16_t)((v.z - mu) * rs * gamma[c + 2] + beta[c + 2]);
    pk.h[3] = (bf16_t)((v.w - mu) * rs * gamma[c + 3] + beta[c + 3]);
    *(uint2*)(w + c) = pk.u;
  }
}

// ---------------------------------------------------------------------------
// Kernel 5: out projection + bias + residual(xn) + transposed store (b,c,f,t).
// ---------------------------------------------------------------------------
__global__ __launch_bounds__(256) void k_out(const bf16_t* __restrict__ on,
                                             const bf16_t* __restrict__ wT,
                                             const float* __restrict__ bias,
                                             const bf16_t* __restrict__ xn,
                                             float* __restrict__ out) {
  __shared__ __align__(16) bf16_t As[2][128 * 48];
  __shared__ __align__(16) bf16_t Bs[2][128 * 48];
  const int tid  = threadIdx.x;
  const int wv   = tid >> 5;
  const int lane = tid & 31;
  const int hi = lane >> 4, nl = lane & 15;
  const int m0 = blockIdx.x * 128;
  const int n0 = blockIdx.y * 128;

  gemm_core(on, wT, m0, n0, As[0], Bs[0], [&](v8f (&acc)[8]) {
    const int token0 = m0 + wv * 16 + 8 * hi;   // 8 consecutive tokens per lane
    const int bb = token0 >> 11, ss0 = token0 & 2047;
#pragma unroll
    for (int ns = 0; ns < 8; ns++) {
      const int n = n0 + ns * 16 + nl;
      const float bv = bias[n];
      float4 v0, v1;
      v0.x = acc[ns][0] + bv + (float)xn[(size_t)(token0 + 0) * CDIM + n];
      v0.y = acc[ns][1] + bv + (float)xn[(size_t)(token0 + 1) * CDIM + n];
      v0.z = acc[ns][2] + bv + (float)xn[(size_t)(token0 + 2) * CDIM + n];
      v0.w = acc[ns][3] + bv + (float)xn[(size_t)(token0 + 3) * CDIM + n];
      v1.x = acc[ns][4] + bv + (float)xn[(size_t)(token0 + 4) * CDIM + n];
      v1.y = acc[ns][5] + bv + (float)xn[(size_t)(token0 + 5) * CDIM + n];
      v1.z = acc[ns][6] + bv + (float)xn[(size_t)(token0 + 6) * CDIM + n];
      v1.w = acc[ns][7] + bv + (float)xn[(size_t)(token0 + 7) * CDIM + n];
      float* dst = out + ((size_t)bb * CDIM + n) * SEQ + ss0;
      *(float4*)(dst)     = v0;
      *(float4*)(dst + 4) = v1;
    }
  });
}

// ---------------------------------------------------------------------------
// Launch
// ---------------------------------------------------------------------------
extern "C" void kernel_launch(void* const* d_in, const int* in_sizes, int n_in,
                              void* d_out, int out_size, void* d_ws, size_t ws_size,
                              hipStream_t stream) {
  const float* x      = (const float*)d_in[0];
  const float* w_qkv  = (const float*)d_in[1];
  const float* b_qkv  = (const float*)d_in[2];
  const float* w_out  = (const float*)d_in[3];
  const float* b_out  = (const float*)d_in[4];
  const float* gamma1 = (const float*)d_in[5];
  const float* beta1  = (const float*)d_in[6];
  const float* gamma2 = (const float*)d_in[7];
  const float* beta2  = (const float*)d_in[8];

  char* ws = (char*)d_ws;
  const size_t SZ_BF = (size_t)NTOK * CDIM * 2;                   // 8 MB
  const size_t SZ_H  = (size_t)BATCH * HEADS * SEQ * DHEAD * 2;   // 8 MB
  bf16_t* xn    = (bf16_t*)(ws);
  bf16_t* qh    = (bf16_t*)(ws + SZ_BF);
  bf16_t* kh    = (bf16_t*)(ws + SZ_BF + SZ_H);
  bf16_t* vTb   = (bf16_t*)(ws + SZ_BF + 2 * SZ_H);
  float*  o     = (float*) (ws + SZ_BF + 3 * SZ_H);               // 16 MB
  char*   p2    = ws + SZ_BF + 3 * SZ_H + (size_t)NTOK * CDIM * 4;
  bf16_t* on    = (bf16_t*)(p2);
  bf16_t* wqkvT = (bf16_t*)(p2 + SZ_BF);
  bf16_t* woutT = (bf16_t*)(p2 + SZ_BF + (size_t)CDIM * N_QKV * 2);

  k_wprep<<<dim3((CDIM * N_QKV) / 256), 256, 0, stream>>>(w_qkv, wqkvT, CDIM, N_QKV);
  k_wprep<<<dim3((CDIM * CDIM) / 256),  256, 0, stream>>>(w_out, woutT, CDIM, CDIM);
  k_ln1 <<<dim3(NTOK / 128),            256, 0, stream>>>(x, gamma1, beta1, xn);
  k_qkv <<<dim3(NTOK / 128, 12),        256, 0, stream>>>(xn, wqkvT, b_qkv, qh, kh, vTb);
  k_attn<<<dim3(SEQ / 128, BATCH * HEADS), 256, 0, stream>>>(qh, kh, vTb, o);
  k_ln2 <<<dim3(NTOK / 256),            256, 0, stream>>>(o, gamma2, beta2, on);
  k_out <<<dim3(NTOK / 128, 4),         256, 0, stream>>>(on, woutT, b_out, xn, (float*)d_out);
}
